// LAESDeepReadout_90469191122970
// MI455X (gfx1250) — matrix operand order repaired and checked by reading, not compiled, not run on hardware
//
#include <hip/hip_runtime.h>
#include <hip/hip_bf16.h>
#include <math.h>

// ---------------------------------------------------------------------------
// LAES recurrence + deep readout for MI455X (gfx1250, wave32, WMMA)
//   h_{t+1} = (x_t - bias) @ A^T + h_t @ B^T    (T=512 steps)
//   out     = tanh(tanh(h W1^T + b1) W2^T + b2) W3^T + b3
//
// Parallel decomposition (chunked associative scan):
//   h_T = sum_t u_t (B^T)^{T-1-t},  u_t = (x_t - bias) A^T
//   16 chunks x 32 steps: local_c computed independently (512 WGs),
//   h = sum_c local_c @ (S^{15-c})^T,  S = B^32 (WMMA power chain).
// Falls back to the monolithic 512-step path if ws_size is too small.
// ---------------------------------------------------------------------------

typedef __bf16 bf16_t;
typedef __attribute__((ext_vector_type(16))) __bf16 v16bf;
typedef __attribute__((ext_vector_type(8)))  __bf16 v8bf;
typedef __attribute__((ext_vector_type(8)))  float  v8f;
typedef __attribute__((ext_vector_type(4)))  int    v4i;

#define T_STEPS 512
#define BATCH   512
#define IN_DIM  128
#define HID     1024
#define NCLS    10
#define NCHUNK  16
#define KTOT    (HID + IN_DIM)     // 1152
#define LSTR    (KTOT + 16)        // 1168 (16B-aligned LDS rows)
#define RSTR    (HID + 16)         // 1040

static __device__ __forceinline__ bf16_t f2bf(float f) { return (bf16_t)f; }
static __device__ __forceinline__ float  bf2f(bf16_t b) { return (float)b; }

// --- scheduling hint: per-kt "DS reads -> VMEM reads -> WMMAs" -------------
#if defined(__gfx1250__) && __has_builtin(__builtin_amdgcn_sched_group_barrier)
// masks: 0x008 = MFMA/WMMA, 0x020 = VMEM read, 0x100 = DS read
#define SCHED_KT_GROUPS()                                          \
    do {                                                           \
        __builtin_amdgcn_sched_group_barrier(0x100, 2, 0);         \
        __builtin_amdgcn_sched_group_barrier(0x020, 16, 0);        \
        __builtin_amdgcn_sched_group_barrier(0x008, 8, 0);         \
    } while (0)
#else
#define SCHED_KT_GROUPS() do {} while (0)
#endif

// --- gfx1250 async global->LDS staging (ASYNCcnt path), guarded ------------
#if defined(__gfx1250__) && __has_builtin(__builtin_amdgcn_global_load_async_to_lds_b128) && __has_builtin(__builtin_amdgcn_s_wait_asynccnt)
#define HAVE_ASYNC_LDS 1
#else
#define HAVE_ASYNC_LDS 0
#endif

// Stage a 16 x HID bf16 tile (row stride HID in global, RSTR in LDS) using
// 256 threads: thread = (row 0..15) x (16 col-groups of 64 bf16 = 8 x 16B).
static __device__ __forceinline__ void stage_tile_16xHID(
    bf16_t* __restrict__ shBase, const bf16_t* __restrict__ gBase, int tid) {
#if HAVE_ASYNC_LDS
    const int r  = tid >> 4;
    const int c0 = (tid & 15) * 64;
    const bf16_t* src = gBase + (size_t)r * HID + c0;
    bf16_t*       dst = shBase + r * RSTR + c0;
    #pragma unroll
    for (int j = 0; j < 8; ++j)
        __builtin_amdgcn_global_load_async_to_lds_b128(
            (v4i*)(src + j * 8), (v4i*)(dst + j * 8), 0, 0);
    __builtin_amdgcn_s_wait_asynccnt(0);
#else
    for (int i = tid; i < 16 * HID; i += 256) {
        int r = i >> 10, c = i & 1023;
        shBase[r * RSTR + c] = gBase[(size_t)r * HID + c];
    }
#endif
}

// ---------------------------------------------------------------------------
// Small prep kernels
// ---------------------------------------------------------------------------
__global__ void build_ba_kernel(const float* __restrict__ A,
                                const float* __restrict__ B,
                                bf16_t* __restrict__ BA) {
    int n = blockIdx.x;
    for (int k = threadIdx.x; k < KTOT; k += blockDim.x) {
        float v = (k < HID) ? B[(size_t)n * HID + k]
                            : A[(size_t)n * IN_DIM + (k - HID)];
        BA[(size_t)n * KTOT + k] = f2bf(v);
    }
}

__global__ void cvt_mat_kernel(const float* __restrict__ W,
                               bf16_t* __restrict__ Wb, int nelem) {
    int i = blockIdx.x * blockDim.x + threadIdx.x;
    if (i < nelem) Wb[i] = f2bf(W[i]);
}

__global__ void identity_bf16_kernel(bf16_t* __restrict__ I) {
    int i = blockIdx.x * blockDim.x + threadIdx.x;
    if (i < HID * HID) {
        int r = i >> 10, c = i & 1023;
        I[i] = (r == c) ? (bf16_t)1.0f : (bf16_t)0.0f;
    }
}

// 32x32 LDS-tile transpose so power-chain B-fragments stay contiguous loads
__global__ void transpose_bf16_kernel(const bf16_t* __restrict__ X,
                                      bf16_t* __restrict__ XT) {
    __shared__ bf16_t tile[32][33];
    const int bx = blockIdx.x * 32, by = blockIdx.y * 32;
    const int tx = threadIdx.x & 31, ty = (threadIdx.x >> 5) * 4;
    #pragma unroll
    for (int j = 0; j < 4; ++j)
        tile[ty + j][tx] = X[(size_t)(by + ty + j) * HID + bx + tx];
    __syncthreads();
    #pragma unroll
    for (int j = 0; j < 4; ++j)
        XT[(size_t)(bx + ty + j) * HID + by + tx] = tile[tx][ty + j];
}

// ---------------------------------------------------------------------------
// Z[m][n] = sum_k X[m][k] * WT[n][k]   (1024x1024x1024, bf16 in/out, f32 acc)
// Used for the B^32 power chain. Grid = 64 WGs (16-row M tiles).
// ---------------------------------------------------------------------------
__global__ __launch_bounds__(256)
void gemm_nt_kernel(const bf16_t* __restrict__ X,
                    const bf16_t* __restrict__ WT,
                    bf16_t* __restrict__ Z) {
    __shared__ bf16_t sh[16 * RSTR];
    const int tid = threadIdx.x, lane = tid & 31, wave = tid >> 5;
    const int m0 = blockIdx.x * 16;

    stage_tile_16xHID(sh, X + (size_t)m0 * HID, tid);
    __syncthreads();

    const int n0w = wave * 128;
    const int am = lane & 15, ak = (lane >> 4) * 16;
    const int bn = lane & 15, bk = (lane >> 4) * 16;

    v8f acc[8] = {};
    #pragma unroll 2
    for (int kt = 0; kt < HID / 32; ++kt) {
        union { v16bf v; v8bf h[2]; } af;
        const bf16_t* arow = &sh[am * RSTR + kt * 32 + ak];
        af.h[0] = *(const v8bf*)(arow);
        af.h[1] = *(const v8bf*)(arow + 8);
        v16bf bw[8];
        #pragma unroll
        for (int nt = 0; nt < 8; ++nt)
            bw[nt] = *(const v16bf*)(WT + (size_t)(n0w + nt * 16 + bn) * HID + kt * 32 + bk);
        #pragma unroll
        for (int nt = 0; nt < 8; ++nt)
            acc[nt] = __builtin_amdgcn_wmma_f32_16x16x32_bf16(
                false, af.v, false, bw[nt], (short)0, acc[nt], false, false);
        SCHED_KT_GROUPS();
    }
    #pragma unroll
    for (int nt = 0; nt < 8; ++nt) {
        const int nc = n0w + nt * 16 + (lane & 15);
        #pragma unroll
        for (int j = 0; j < 8; ++j) {
            const int m = j + ((lane >> 4) << 3);
            Z[(size_t)(m0 + m) * HID + nc] = f2bf(acc[nt][j]);
        }
    }
}

// ---------------------------------------------------------------------------
// Chunked recurrence: grid (BATCH/16, nchunks), 256 threads (8 waves).
// Chunk blockIdx.y runs steps [y*nsteps, (y+1)*nsteps) from h=0 and writes
// its local partial h to dst[y]. nchunks=1,nsteps=512 == monolithic path.
// ---------------------------------------------------------------------------
__global__ __launch_bounds__(256)
void laes_recurrence_kernel(const float* __restrict__ x,
                            const float* __restrict__ bias,
                            const bf16_t* __restrict__ BA,
                            bf16_t* __restrict__ dst,
                            int nsteps) {
    __shared__ bf16_t sop[16 * LSTR];   // ~37.4 KB

    const int tid  = threadIdx.x;
    const int lane = tid & 31;
    const int wave = tid >> 5;
    const int b0   = blockIdx.x * 16;
    const int t0   = blockIdx.y * nsteps;

    const int xr = tid >> 4;            // x staging: 8 cols/thread
    const int xc = (tid & 15) * 8;
    float biasv[8];
    #pragma unroll
    for (int j = 0; j < 8; ++j) biasv[j] = bias[xc + j];

    // init: h0 = 0, stage (x[t0]-bias)
    for (int i = tid; i < 16 * HID; i += 256) {
        int r = i >> 10, c = i & 1023;
        sop[r * LSTR + c] = (bf16_t)0.0f;
    }
    {
        const float* xp = x + ((size_t)t0 * BATCH + (b0 + xr)) * IN_DIM + xc;
        #pragma unroll
        for (int j = 0; j < 8; ++j)
            sop[xr * LSTR + HID + xc + j] = f2bf(xp[j] - biasv[j]);
    }
    __syncthreads();

    const int n0w = wave * 128;
    const int am = lane & 15, ak = (lane >> 4) * 16;  // A-frag (LDS)
    const int bn = lane & 15, bk = (lane >> 4) * 16;  // B-frag (global)

    #pragma unroll 1
    for (int s = 0; s < nsteps; ++s) {
        const int t  = t0 + s;
        const int tn = (t + 1 < T_STEPS) ? (t + 1) : (T_STEPS - 1);
        const int tp = (t + 2 < T_STEPS) ? (t + 2) : (T_STEPS - 1);
        const float* xp = x + ((size_t)tn * BATCH + (b0 + xr)) * IN_DIM + xc;
        __builtin_prefetch(x + ((size_t)tp * BATCH + (b0 + xr)) * IN_DIM + xc, 0, 1);
        float xv[8];
        #pragma unroll
        for (int j = 0; j < 8; ++j) xv[j] = xp[j];

        v8f acc[8] = {};
        #pragma unroll 2
        for (int kt = 0; kt < KTOT / 32; ++kt) {
            union { v16bf v; v8bf h[2]; } af;
            const bf16_t* arow = &sop[am * LSTR + kt * 32 + ak];
            af.h[0] = *(const v8bf*)(arow);
            af.h[1] = *(const v8bf*)(arow + 8);
            v16bf bw[8];
            #pragma unroll
            for (int nt = 0; nt < 8; ++nt)
                bw[nt] = *(const v16bf*)(BA + (size_t)(n0w + nt * 16 + bn) * KTOT + kt * 32 + bk);
            #pragma unroll
            for (int nt = 0; nt < 8; ++nt)
                acc[nt] = __builtin_amdgcn_wmma_f32_16x16x32_bf16(
                    false, af.v, false, bw[nt], (short)0, acc[nt], false, false);
            SCHED_KT_GROUPS();
        }

        __syncthreads();   // all reads of sop done

        #pragma unroll
        for (int nt = 0; nt < 8; ++nt) {
            const int nc = n0w + nt * 16 + (lane & 15);
            #pragma unroll
            for (int j = 0; j < 8; ++j) {
                const int m = j + ((lane >> 4) << 3);
                sop[m * LSTR + nc] = f2bf(acc[nt][j]);
            }
        }
        #pragma unroll
        for (int j = 0; j < 8; ++j)
            sop[xr * LSTR + HID + xc + j] = f2bf(xv[j] - biasv[j]);

        __syncthreads();
    }

    // Export this chunk's local partial h
    bf16_t* d = dst + (size_t)blockIdx.y * BATCH * HID;
    for (int i = tid; i < 16 * HID; i += 256) {
        int r = i >> 10, c = i & 1023;
        d[(size_t)(b0 + r) * HID + c] = sop[r * LSTR + c];
    }
}

// ---------------------------------------------------------------------------
// Combine: hb[m][n] = sum_c sum_k local_c[m][k] * Ppow[15-c][n][k]
// Ppow[j] = (B^32)^j row-major (Ppow[0] = I). Grid = 32 WGs.
// ---------------------------------------------------------------------------
__global__ __launch_bounds__(256)
void combine_kernel(const bf16_t* __restrict__ locals,  // [16][512][1024]
                    const bf16_t* __restrict__ Ppow,    // [16][1024][1024]
                    bf16_t* __restrict__ hb) {
    __shared__ bf16_t sh[16 * RSTR];
    const int tid = threadIdx.x, lane = tid & 31, wave = tid >> 5;
    const int b0 = blockIdx.x * 16;

    const int n0w = wave * 128;
    const int am = lane & 15, ak = (lane >> 4) * 16;
    const int bn = lane & 15, bk = (lane >> 4) * 16;

    v8f acc[8] = {};
    #pragma unroll 1
    for (int c = 0; c < NCHUNK; ++c) {
        stage_tile_16xHID(sh, locals + (size_t)c * BATCH * HID + (size_t)b0 * HID, tid);
        __syncthreads();

        const bf16_t* W = Ppow + (size_t)(NCHUNK - 1 - c) * HID * HID;
        #pragma unroll 2
        for (int kt = 0; kt < HID / 32; ++kt) {
            union { v16bf v; v8bf h[2]; } af;
            const bf16_t* arow = &sh[am * RSTR + kt * 32 + ak];
            af.h[0] = *(const v8bf*)(arow);
            af.h[1] = *(const v8bf*)(arow + 8);
            v16bf bw[8];
            #pragma unroll
            for (int nt = 0; nt < 8; ++nt)
                bw[nt] = *(const v16bf*)(W + (size_t)(n0w + nt * 16 + bn) * HID + kt * 32 + bk);
            #pragma unroll
            for (int nt = 0; nt < 8; ++nt)
                acc[nt] = __builtin_amdgcn_wmma_f32_16x16x32_bf16(
                    false, af.v, false, bw[nt], (short)0, acc[nt], false, false);
            SCHED_KT_GROUPS();
        }
        __syncthreads();   // done reading sh before next chunk restages
    }

    #pragma unroll
    for (int nt = 0; nt < 8; ++nt) {
        const int nc = n0w + nt * 16 + (lane & 15);
        #pragma unroll
        for (int j = 0; j < 8; ++j) {
            const int m = j + ((lane >> 4) << 3);
            hb[(size_t)(b0 + m) * HID + nc] = f2bf(acc[nt][j]);
        }
    }
}

// ---------------------------------------------------------------------------
// Readout: two WMMA+tanh layers (K=1024) and a scalar 10-class final layer.
// ---------------------------------------------------------------------------
__global__ __launch_bounds__(256)
void laes_readout_kernel(const bf16_t* __restrict__ hb,
                         const bf16_t* __restrict__ W1b, const float* __restrict__ b1,
                         const bf16_t* __restrict__ W2b, const float* __restrict__ b2,
                         const float* __restrict__ W3,  const float* __restrict__ b3,
                         float* __restrict__ out) {
    __shared__ bf16_t sh[16 * RSTR];
    const int tid = threadIdx.x, lane = tid & 31, wave = tid >> 5;
    const int b0 = blockIdx.x * 16;

    stage_tile_16xHID(sh, hb + (size_t)b0 * HID, tid);
    __syncthreads();

    const int n0w = wave * 128;
    const int am = lane & 15, ak = (lane >> 4) * 16;
    const int bn = lane & 15, bk = (lane >> 4) * 16;

    #pragma unroll 1
    for (int layer = 0; layer < 2; ++layer) {
        const bf16_t* Wb = layer ? (const bf16_t*)W2b : (const bf16_t*)W1b;
        const float*  bv = layer ? b2 : b1;

        v8f acc[8] = {};
        #pragma unroll 2
        for (int kt = 0; kt < HID / 32; ++kt) {
            union { v16bf v; v8bf h[2]; } af;
            const bf16_t* arow = &sh[am * RSTR + kt * 32 + ak];
            af.h[0] = *(const v8bf*)(arow);
            af.h[1] = *(const v8bf*)(arow + 8);
            v16bf bw[8];
            #pragma unroll
            for (int nt = 0; nt < 8; ++nt)
                bw[nt] = *(const v16bf*)(Wb + (size_t)(n0w + nt * 16 + bn) * HID + kt * 32 + bk);
            #pragma unroll
            for (int nt = 0; nt < 8; ++nt)
                acc[nt] = __builtin_amdgcn_wmma_f32_16x16x32_bf16(
                    false, af.v, false, bw[nt], (short)0, acc[nt], false, false);
            SCHED_KT_GROUPS();
        }
        __syncthreads();
        #pragma unroll
        for (int nt = 0; nt < 8; ++nt) {
            const int nc = n0w + nt * 16 + (lane & 15);
            const float bb = bv[nc];
            #pragma unroll
            for (int j = 0; j < 8; ++j) {
                const int m = j + ((lane >> 4) << 3);
                sh[m * RSTR + nc] = f2bf(tanhf(acc[nt][j] + bb));
            }
        }
        __syncthreads();
    }

    if (tid < 16 * NCLS) {
        const int m = tid / NCLS, c = tid % NCLS;
        float s = b3[c];
        const float* w = W3 + (size_t)c * HID;
        for (int k = 0; k < HID; ++k)
            s += bf2f(sh[m * RSTR + k]) * w[k];
        out[(size_t)(b0 + m) * NCLS + c] = s;
    }
}

// ---------------------------------------------------------------------------
// Workspace layout (bytes)
// ---------------------------------------------------------------------------
#define OFF_BA   ((size_t)0)             // 2,359,296
#define OFF_W1   ((size_t)2359296)       // 2,097,152
#define OFF_W2   ((size_t)4456448)       // 2,097,152
#define OFF_HB   ((size_t)6553600)       // 1,048,576
#define NEED_BASE ((size_t)7602176)
#define OFF_PA   ((size_t)7602176)       // 2,097,152
#define OFF_PB   ((size_t)9699328)       // 2,097,152
#define OFF_PT   ((size_t)11796480)      // 2,097,152
#define OFF_PPOW ((size_t)13893632)      // 16 x 2,097,152
#define OFF_LOC  ((size_t)47448064)      // 16 x 1,048,576
#define NEED_SCAN ((size_t)64225280)

extern "C" void kernel_launch(void* const* d_in, const int* in_sizes, int n_in,
                              void* d_out, int out_size, void* d_ws, size_t ws_size,
                              hipStream_t stream) {
    const float* x    = (const float*)d_in[0];
    const float* A    = (const float*)d_in[1];
    const float* B    = (const float*)d_in[2];
    const float* bias = (const float*)d_in[3];
    const float* W1   = (const float*)d_in[4];
    const float* b1   = (const float*)d_in[5];
    const float* W2   = (const float*)d_in[6];
    const float* b2   = (const float*)d_in[7];
    const float* W3   = (const float*)d_in[8];
    const float* b3   = (const float*)d_in[9];
    float* out = (float*)d_out;

    char* ws = (char*)d_ws;
    bf16_t* BA   = (bf16_t*)(ws + OFF_BA);
    bf16_t* W1b  = (bf16_t*)(ws + OFF_W1);
    bf16_t* W2b  = (bf16_t*)(ws + OFF_W2);
    bf16_t* hb   = (bf16_t*)(ws + OFF_HB);

    build_ba_kernel<<<HID, 256, 0, stream>>>(A, B, BA);
    cvt_mat_kernel<<<(HID * HID + 255) / 256, 256, 0, stream>>>(W1, W1b, HID * HID);
    cvt_mat_kernel<<<(HID * HID + 255) / 256, 256, 0, stream>>>(W2, W2b, HID * HID);

    if (ws_size >= NEED_SCAN) {
        // ---- chunked associative scan: 512 WGs on the recurrence ----
        bf16_t* PA   = (bf16_t*)(ws + OFF_PA);
        bf16_t* PB   = (bf16_t*)(ws + OFF_PB);
        bf16_t* PT   = (bf16_t*)(ws + OFF_PT);
        bf16_t* Ppow = (bf16_t*)(ws + OFF_PPOW);
        bf16_t* locs = (bf16_t*)(ws + OFF_LOC);

        // S = B^32 by five WMMA squarings (ping-pong PA/PB)
        cvt_mat_kernel<<<(HID * HID + 255) / 256, 256, 0, stream>>>(B, PA, HID * HID);
        bf16_t* pa = PA; bf16_t* pb = PB;
        dim3 tg(HID / 32, HID / 32);
        for (int i = 0; i < 5; ++i) {
            transpose_bf16_kernel<<<tg, 256, 0, stream>>>(pa, PT);
            gemm_nt_kernel<<<HID / 16, 256, 0, stream>>>(pa, PT, pb);
            bf16_t* t = pa; pa = pb; pb = t;
        }
        // Ppow[0] = I, Ppow[1] = S, Ppow[j] = Ppow[j-1] @ S
        identity_bf16_kernel<<<(HID * HID + 255) / 256, 256, 0, stream>>>(Ppow);
        (void)hipMemcpyAsync(Ppow + (size_t)HID * HID, pa,
                             (size_t)HID * HID * sizeof(bf16_t),
                             hipMemcpyDeviceToDevice, stream);
        transpose_bf16_kernel<<<tg, 256, 0, stream>>>(pa, PT);   // S^T
        for (int j = 2; j < NCHUNK; ++j) {
            gemm_nt_kernel<<<HID / 16, 256, 0, stream>>>(
                Ppow + (size_t)(j - 1) * HID * HID, PT,
                Ppow + (size_t)j * HID * HID);
        }

        // 16 independent 32-step chunk recurrences, then WMMA combine
        dim3 grid(BATCH / 16, NCHUNK);
        laes_recurrence_kernel<<<grid, 256, 0, stream>>>(x, bias, BA, locs,
                                                         T_STEPS / NCHUNK);
        combine_kernel<<<BATCH / 16, 256, 0, stream>>>(locs, Ppow, hb);
    } else {
        // ---- fallback: monolithic 512-step recurrence (32 WGs) ----
        dim3 grid(BATCH / 16, 1);
        laes_recurrence_kernel<<<grid, 256, 0, stream>>>(x, bias, BA, hb, T_STEPS);
    }

    laes_readout_kernel<<<BATCH / 16, 256, 0, stream>>>(hb, W1b, b1, W2b, b2,
                                                        W3, b3, out);
}
